// Model_37847251812517
// MI455X (gfx1250) — matrix operand order repaired and checked by reading, not compiled, not run on hardware
//
#include <hip/hip_runtime.h>
#include <hip/hip_bf16.h>
#include <math.h>

// Model dims
#define B_  8
#define S_  512
#define E_  768
#define NH_ 12
#define DH_ 64
#define FF_ 3072
#define L_  6
#define C_  10
#define MTOK (B_ * S_)          // 4096 token rows
#define HB   (B_ * NH_)         // 96 attention heads total
#define HSTR (S_ * DH_)         // 32768 elements per head block

#define BK 32                    // WMMA K per step

typedef __bf16 v16bf __attribute__((ext_vector_type(16)));
typedef float  v8f   __attribute__((ext_vector_type(8)));
typedef int    v4i   __attribute__((ext_vector_type(4)));

union FragBF { unsigned int u[8]; v16bf v; };

// --------------------------------------------------------------------------
// CDNA5 async global->LDS copy (ASYNCcnt-tracked DMA), guarded so the file
// still compiles on toolchains without the builtin (falls back to batched
// register staging). Builtin signature (from compiler diagnostic):
//   (v4i AS1* src, v4i AS3* dst, imm offset, imm cpol)
// --------------------------------------------------------------------------
#if defined(__has_builtin)
# if __has_builtin(__builtin_amdgcn_global_load_async_to_lds_b128)
#  define USE_ASYNC_LDS 1
# endif
#endif
#ifndef USE_ASYNC_LDS
# define USE_ASYNC_LDS 0
#endif

__device__ __forceinline__ void stage16_async(const __hip_bfloat16* g,
                                              unsigned short* l)
{
#if USE_ASYNC_LDS
    __builtin_amdgcn_global_load_async_to_lds_b128(
        (__attribute__((address_space(1))) v4i*)const_cast<__hip_bfloat16*>(g),
        (__attribute__((address_space(3))) v4i*)l, 0, 0);
#else
    (void)g; (void)l;
#endif
}

__device__ __forceinline__ void stage_wait_async()
{
#if USE_ASYNC_LDS
# if __has_builtin(__builtin_amdgcn_s_wait_asynccnt)
    __builtin_amdgcn_s_wait_asynccnt(0);
# else
    asm volatile("s_wait_asynccnt 0x0" ::: "memory");
# endif
#endif
}

// ---------------------------------------------------------------------------
// bf16 WMMA GEMM, B supplied as [N,K] row-major ("TN" layout):
//   C[M,N] = alpha * A[M,K] @ B[N,K]^T + bias, opt ReLU, opt bf16 out, batched.
// Double-buffered LDS; staging of tile k+1 overlaps WMMA compute of tile k.
// ---------------------------------------------------------------------------
template <int TBM, int TBN, int BLOCK>
__global__ __launch_bounds__(BLOCK)
void gemm_tn(const __hip_bfloat16* __restrict__ A, int lda, long long sA,
             const __hip_bfloat16* __restrict__ Bm, int ldb, long long sB,
             void* __restrict__ Cm, int ldc, long long sC,
             int K, float alpha, const float* __restrict__ bias,
             int relu, int out_bf16)
{
    constexpr int LDTS   = 40;               // ushort row stride: 80B, 16B aligned
    constexpr int WAVES  = BLOCK / 32;
    constexpr int ROWS_W = TBM / 32;         // wave grid rows
    constexpr int COLS_W = WAVES / ROWS_W;   // wave grid cols
    constexpr int WN     = TBN / COLS_W;     // wave tile N
    constexpr int NFR    = WN / 16;          // B fragments per wave
    static_assert(ROWS_W * COLS_W == WAVES, "wave grid");
    constexpr int AIT = (TBM * 4) / BLOCK;   // 16B chunks per thread for A tile
    constexpr int BIT = (TBN * 4) / BLOCK;

    __shared__ unsigned short At[2][TBM * LDTS];
    __shared__ unsigned short Bt[2][TBN * LDTS];

    const int bz = blockIdx.z;
    A  += (size_t)bz * sA;
    Bm += (size_t)bz * sB;

    const int m0    = blockIdx.y * TBM;
    const int n0    = blockIdx.x * TBN;
    const int tid   = threadIdx.x;
    const int lane  = tid & 31;
    const int wave  = tid >> 5;
    const int wm    = (wave % ROWS_W) * 32;
    const int wn    = (wave / ROWS_W) * WN;
    const int row16 = lane & 15;
    const int hl    = lane >> 4;

    // Per-thread staging assignments (hoisted out of K loop; bump by BK)
    const __hip_bfloat16* aSrc[AIT]; int aOff[AIT];
    const __hip_bfloat16* bSrc[BIT]; int bOff[BIT];
    #pragma unroll
    for (int it = 0; it < AIT; ++it) {
        int c = tid + it * BLOCK, r = c >> 2, q = c & 3;
        aSrc[it] = A + (size_t)(m0 + r) * lda + q * 8;
        aOff[it] = r * LDTS + q * 8;
    }
    #pragma unroll
    for (int it = 0; it < BIT; ++it) {
        int c = tid + it * BLOCK, r = c >> 2, q = c & 3;
        bSrc[it] = Bm + (size_t)(n0 + r) * ldb + q * 8;
        bOff[it] = r * LDTS + q * 8;
    }

    v8f acc[2][NFR];
    #pragma unroll
    for (int i = 0; i < 2; ++i)
        #pragma unroll
        for (int j = 0; j < NFR; ++j)
            #pragma unroll
            for (int r = 0; r < 8; ++r) acc[i][j][r] = 0.0f;

    // ---- prologue: stage tile k0 = 0 into buffer 0 ----
#if USE_ASYNC_LDS
    #pragma unroll
    for (int it = 0; it < AIT; ++it) { stage16_async(aSrc[it], &At[0][aOff[it]]); aSrc[it] += BK; }
    #pragma unroll
    for (int it = 0; it < BIT; ++it) { stage16_async(bSrc[it], &Bt[0][bOff[it]]); bSrc[it] += BK; }
    stage_wait_async();
#else
    {
        uint4 ra[AIT], rb[BIT];
        #pragma unroll
        for (int it = 0; it < AIT; ++it) { ra[it] = *(const uint4*)aSrc[it]; aSrc[it] += BK; }
        #pragma unroll
        for (int it = 0; it < BIT; ++it) { rb[it] = *(const uint4*)bSrc[it]; bSrc[it] += BK; }
        #pragma unroll
        for (int it = 0; it < AIT; ++it) *(uint4*)&At[0][aOff[it]] = ra[it];
        #pragma unroll
        for (int it = 0; it < BIT; ++it) *(uint4*)&Bt[0][bOff[it]] = rb[it];
    }
#endif
    __syncthreads();

    int cur = 0;
    for (int k0 = 0; k0 < K; k0 += BK) {
        const int  nxt  = cur ^ 1;
        const bool more = (k0 + BK) < K;

#if USE_ASYNC_LDS
        if (more) {
            #pragma unroll
            for (int it = 0; it < AIT; ++it) { stage16_async(aSrc[it], &At[nxt][aOff[it]]); aSrc[it] += BK; }
            #pragma unroll
            for (int it = 0; it < BIT; ++it) { stage16_async(bSrc[it], &Bt[nxt][bOff[it]]); bSrc[it] += BK; }
        }
#else
        uint4 ra[AIT], rb[BIT];
        if (more) {
            #pragma unroll
            for (int it = 0; it < AIT; ++it) { ra[it] = *(const uint4*)aSrc[it]; aSrc[it] += BK; }
            #pragma unroll
            for (int it = 0; it < BIT; ++it) { rb[it] = *(const uint4*)bSrc[it]; bSrc[it] += BK; }
        }
#endif

        // ---- compute on current buffer ----
        const unsigned short* Ab = At[cur];
        const unsigned short* Bb = Bt[cur];
        FragBF fa[2], fb[NFR];
        #pragma unroll
        for (int i = 0; i < 2; ++i) {
            int ar = wm + i * 16 + row16;
            // A layout: vgpr v -> K = {0,2,4,6 | 16,18,20,22} + 8*half
            *(uint4*)&fa[i].u[0] = *(const uint4*)&Ab[ar * LDTS + 8 * hl];
            *(uint4*)&fa[i].u[4] = *(const uint4*)&Ab[ar * LDTS + 16 + 8 * hl];
        }
        #pragma unroll
        for (int j = 0; j < NFR; ++j) {
            int br = wn + j * 16 + row16;
            // B layout: vgpr v -> K = {0,2,4,6 | 16,18,20,22} + 8*half
            *(uint4*)&fb[j].u[0] = *(const uint4*)&Bb[br * LDTS + 8 * hl];
            *(uint4*)&fb[j].u[4] = *(const uint4*)&Bb[br * LDTS + 16 + 8 * hl];
        }
        #pragma unroll
        for (int i = 0; i < 2; ++i)
            #pragma unroll
            for (int j = 0; j < NFR; ++j)
                acc[i][j] = __builtin_amdgcn_wmma_f32_16x16x32_bf16(
                    false, fa[i].v, false, fb[j].v, (short)0, acc[i][j],
                    false, false);

#if USE_ASYNC_LDS
        stage_wait_async();     // my async writes to LDS[nxt] complete
#else
        if (more) {
            #pragma unroll
            for (int it = 0; it < AIT; ++it) *(uint4*)&At[nxt][aOff[it]] = ra[it];
            #pragma unroll
            for (int it = 0; it < BIT; ++it) *(uint4*)&Bt[nxt][bOff[it]] = rb[it];
        }
#endif
        __syncthreads();        // everyone done reading cur / writing nxt
        cur = nxt;
    }

    // ---- epilogue: D element (vgpr r, lane) = (M = r + 8*hl, N = lane&15) ----
    #pragma unroll
    for (int i = 0; i < 2; ++i) {
        #pragma unroll
        for (int j = 0; j < NFR; ++j) {
            int nIdx = n0 + wn + j * 16 + row16;
            float bval = bias ? bias[nIdx] : 0.0f;
            #pragma unroll
            for (int r = 0; r < 8; ++r) {
                int mIdx = m0 + wm + i * 16 + r + 8 * hl;
                float vv = acc[i][j][r] * alpha + bval;
                if (relu) vv = fmaxf(vv, 0.0f);
                size_t cidx = (size_t)bz * sC + (size_t)mIdx * ldc + nIdx;
                if (out_bf16)
                    ((__hip_bfloat16*)Cm)[cidx] = __float2bfloat16(vv);
                else
                    ((float*)Cm)[cidx] = vv;
            }
        }
    }
}

// ---------------------------------------------------------------------------
// Weight transpose + cast: W[K,N] f32 -> Wt[N,K] bf16 (32x32 LDS tiles)
// ---------------------------------------------------------------------------
__global__ __launch_bounds__(256)
void transpose_cast_w(const float* __restrict__ W, __hip_bfloat16* __restrict__ Wt,
                      int K, int N)
{
    __shared__ float t[32][33];
    const int n0 = blockIdx.x * 32;
    const int k0 = blockIdx.y * 32;
    const int tx = threadIdx.x & 31;
    const int ty = threadIdx.x >> 5;
    #pragma unroll
    for (int r = 0; r < 32; r += 8)
        t[ty + r][tx] = W[(size_t)(k0 + ty + r) * N + n0 + tx];
    __syncthreads();
    #pragma unroll
    for (int r = 0; r < 32; r += 8)
        Wt[(size_t)(n0 + ty + r) * K + k0 + tx] = __float2bfloat16(t[tx][ty + r]);
}

// ---------------------------------------------------------------------------
// Per-head V transpose (bf16): V[96][512][64] -> VT[96][64][512]
// ---------------------------------------------------------------------------
__global__ __launch_bounds__(256)
void transpose_v_head(const __hip_bfloat16* __restrict__ V,
                      __hip_bfloat16* __restrict__ VT)
{
    __shared__ __hip_bfloat16 t[32][33];
    const int g  = blockIdx.z;
    const int d0 = blockIdx.x * 32;   // DH/32 = 2
    const int s0 = blockIdx.y * 32;   // S/32  = 16
    const __hip_bfloat16* src = V  + (size_t)g * HSTR;
    __hip_bfloat16*       dst = VT + (size_t)g * HSTR;
    const int tx = threadIdx.x & 31;
    const int ty = threadIdx.x >> 5;
    #pragma unroll
    for (int r = 0; r < 32; r += 8)
        t[ty + r][tx] = src[(size_t)(s0 + ty + r) * DH_ + d0 + tx];
    __syncthreads();
    #pragma unroll
    for (int r = 0; r < 32; r += 8)
        dst[(size_t)(d0 + ty + r) * S_ + s0 + tx] = t[tx][ty + r];
}

// ---------------------------------------------------------------------------
// Embedding gather: h[row] = emb[x[row]] (f32 + bf16 copies)
// ---------------------------------------------------------------------------
__global__ __launch_bounds__(256)
void embed_gather(const int* __restrict__ x, const float* __restrict__ emb,
                  float* __restrict__ h, __hip_bfloat16* __restrict__ hb)
{
    int row = blockIdx.x;
    int tok = x[row];
    const float* src = emb + (size_t)tok * E_;
    size_t base = (size_t)row * E_;
    for (int e = threadIdx.x; e < E_; e += 256) {
        float v = src[e];
        h[base + e]  = v;
        hb[base + e] = __float2bfloat16(v);
    }
}

// ---------------------------------------------------------------------------
// Row softmax, fixed 512 cols, 2 elements/thread -> bf16
// ---------------------------------------------------------------------------
__global__ __launch_bounds__(256)
void softmax_rows(const float* __restrict__ Sc, __hip_bfloat16* __restrict__ P)
{
    __shared__ float red[256];
    const int row = blockIdx.x;
    const float* x = Sc + (size_t)row * S_;
    __hip_bfloat16* y = P + (size_t)row * S_;
    const int tid = threadIdx.x;

    float v0 = x[tid];
    float v1 = x[tid + 256];

    red[tid] = fmaxf(v0, v1); __syncthreads();
    for (int s = 128; s > 0; s >>= 1) {
        if (tid < s) red[tid] = fmaxf(red[tid], red[tid + s]);
        __syncthreads();
    }
    float m = red[0]; __syncthreads();

    float e0 = __expf(v0 - m);
    float e1 = __expf(v1 - m);
    red[tid] = e0 + e1; __syncthreads();
    for (int s = 128; s > 0; s >>= 1) {
        if (tid < s) red[tid] += red[tid + s];
        __syncthreads();
    }
    float inv = 1.0f / red[0];

    y[tid]       = __float2bfloat16(e0 * inv);
    y[tid + 256] = __float2bfloat16(e1 * inv);
}

// ---------------------------------------------------------------------------
// h = LayerNorm(a + h) * g + b ; writes f32 h and bf16 copy. Row = 768.
// ---------------------------------------------------------------------------
__global__ __launch_bounds__(256)
void residual_layernorm(const float* __restrict__ a, float* __restrict__ h,
                        __hip_bfloat16* __restrict__ hb,
                        const float* __restrict__ g, const float* __restrict__ b)
{
    __shared__ float red[256];
    const int row = blockIdx.x;
    const size_t base = (size_t)row * E_;
    const int tid = threadIdx.x;

    float t[3];
    float s = 0.0f;
    #pragma unroll
    for (int i = 0; i < 3; ++i) {
        int c = tid + 256 * i;
        t[i] = a[base + c] + h[base + c];
        s += t[i];
    }
    red[tid] = s; __syncthreads();
    for (int st = 128; st > 0; st >>= 1) {
        if (tid < st) red[tid] += red[tid + st];
        __syncthreads();
    }
    float mean = red[0] * (1.0f / E_); __syncthreads();

    float ss = 0.0f;
    #pragma unroll
    for (int i = 0; i < 3; ++i) { float d = t[i] - mean; ss += d * d; }
    red[tid] = ss; __syncthreads();
    for (int st = 128; st > 0; st >>= 1) {
        if (tid < st) red[tid] += red[tid + st];
        __syncthreads();
    }
    float inv = rsqrtf(red[0] * (1.0f / E_) + 1e-5f);

    #pragma unroll
    for (int i = 0; i < 3; ++i) {
        int c = tid + 256 * i;
        float o = (t[i] - mean) * inv * g[c] + b[c];
        h[base + c]  = o;
        hb[base + c] = __float2bfloat16(o);
    }
}

// ---------------------------------------------------------------------------
// Classifier: out[b, :] = h[b, 0:393216] @ Wc[393216, 10] + bc
// ---------------------------------------------------------------------------
__global__ __launch_bounds__(256)
void classifier(const float* __restrict__ h, const float* __restrict__ Wc,
                const float* __restrict__ bc, float* __restrict__ out)
{
    __shared__ float red[256];
    const int b = blockIdx.x;
    const float* hb = h + (size_t)b * (S_ * E_);
    const int tid = threadIdx.x;

    float acc[C_];
    #pragma unroll
    for (int c = 0; c < C_; ++c) acc[c] = 0.0f;

    for (int i = tid; i < S_ * E_; i += 256) {
        float hv = hb[i];
        const float* w = Wc + (size_t)i * C_;
        #pragma unroll
        for (int c = 0; c < C_; ++c) acc[c] += hv * w[c];
    }
    for (int c = 0; c < C_; ++c) {
        red[tid] = acc[c]; __syncthreads();
        for (int s = 128; s > 0; s >>= 1) {
            if (tid < s) red[tid] += red[tid + s];
            __syncthreads();
        }
        if (tid == 0) out[b * C_ + c] = red[0] + bc[c];
        __syncthreads();
    }
}

// ---------------------------------------------------------------------------
// Host orchestration
// ---------------------------------------------------------------------------
extern "C" void kernel_launch(void* const* d_in, const int* in_sizes, int n_in,
                              void* d_out, int out_size, void* d_ws, size_t ws_size,
                              hipStream_t stream)
{
    const int*   x    = (const int*)  d_in[0];
    const float* emb  = (const float*)d_in[1];
    const float* Wq   = (const float*)d_in[2];
    const float* bq   = (const float*)d_in[3];
    const float* Wk   = (const float*)d_in[4];
    const float* bk   = (const float*)d_in[5];
    const float* Wv   = (const float*)d_in[6];
    const float* bv   = (const float*)d_in[7];
    const float* Wo   = (const float*)d_in[8];
    const float* bo   = (const float*)d_in[9];
    const float* ln1g = (const float*)d_in[10];
    const float* ln1b = (const float*)d_in[11];
    const float* W1   = (const float*)d_in[12];
    const float* b1   = (const float*)d_in[13];
    const float* W2   = (const float*)d_in[14];
    const float* b2   = (const float*)d_in[15];
    const float* ln2g = (const float*)d_in[16];
    const float* ln2b = (const float*)d_in[17];
    const float* Wc   = (const float*)d_in[18];
    const float* bc   = (const float*)d_in[19];
    float* out = (float*)d_out;

    char* ws = (char*)d_ws;
    size_t off = 0;
    auto alloc = [&](size_t bytes) -> void* {
        void* p = ws + off;
        off = (off + bytes + 255) & ~(size_t)255;
        return p;
    };

    float*          h_f32   = (float*)         alloc((size_t)MTOK * E_ * 4);
    __hip_bfloat16* h_bf    = (__hip_bfloat16*)alloc((size_t)MTOK * E_ * 2);
    __hip_bfloat16* q_bf    = (__hip_bfloat16*)alloc((size_t)MTOK * E_ * 2);
    __hip_bfloat16* k_bf    = (__hip_bfloat16*)alloc((size_t)MTOK * E_ * 2);
    __hip_bfloat16* v_bf    = (__hip_bfloat16*)alloc((size_t)MTOK * E_ * 2);
    __hip_bfloat16* vT_bf   = (__hip_bfloat16*)alloc((size_t)MTOK * E_ * 2);
    float*          scores  = (float*)         alloc((size_t)HB * S_ * S_ * 4);
    __hip_bfloat16* attn_bf = (__hip_bfloat16*)alloc((size_t)HB * S_ * S_ * 2);
    __hip_bfloat16* ctx_bf  = (__hip_bfloat16*)alloc((size_t)MTOK * E_ * 2);
    float*          tmp_f32 = (float*)         alloc((size_t)MTOK * E_ * 4);
    __hip_bfloat16* ffn1_bf = (__hip_bfloat16*)alloc((size_t)MTOK * FF_ * 2);
    __hip_bfloat16* wbuf    = (__hip_bfloat16*)alloc((size_t)(4 * E_ * E_ + 2 * E_ * FF_) * 2);

    // Transposed ([N,K]) bf16 weights for this layer
    __hip_bfloat16* wq_t = wbuf;
    __hip_bfloat16* wk_t = wq_t + (size_t)E_ * E_;
    __hip_bfloat16* wv_t = wk_t + (size_t)E_ * E_;
    __hip_bfloat16* wo_t = wv_t + (size_t)E_ * E_;
    __hip_bfloat16* w1_t = wo_t + (size_t)E_ * E_;   // [FF, E]
    __hip_bfloat16* w2_t = w1_t + (size_t)E_ * FF_;  // [E, FF]

    auto tcast = [&](const float* W, __hip_bfloat16* Wt, int K, int N) {
        transpose_cast_w<<<dim3(N / 32, K / 32, 1), 256, 0, stream>>>(W, Wt, K, N);
    };
    auto gemm_big = [&](const __hip_bfloat16* A, int lda, long long sA,
                        const __hip_bfloat16* Bm, int ldb, long long sB,
                        void* Cm, int ldc, long long sC,
                        int M, int N, int K, int batch, float alpha,
                        const float* bias, int relu, int out_bf16) {
        dim3 grid(N / 128, M / 128, batch);
        gemm_tn<128, 128, 256><<<grid, 256, 0, stream>>>(
            A, lda, sA, Bm, ldb, sB, Cm, ldc, sC, K, alpha, bias, relu, out_bf16);
    };
    auto gemm_sm = [&](const __hip_bfloat16* A, int lda, long long sA,
                       const __hip_bfloat16* Bm, int ldb, long long sB,
                       void* Cm, int ldc, long long sC,
                       int M, int N, int K, int batch, float alpha,
                       const float* bias, int relu, int out_bf16) {
        dim3 grid(N / 64, M / 64, batch);
        gemm_tn<64, 64, 128><<<grid, 128, 0, stream>>>(
            A, lda, sA, Bm, ldb, sB, Cm, ldc, sC, K, alpha, bias, relu, out_bf16);
    };

    // Embedding
    embed_gather<<<MTOK, 256, 0, stream>>>(x, emb, h_f32, h_bf);

    const float attn_scale = 0.125f; // 1/sqrt(64)

    for (int l = 0; l < L_; ++l) {
        // Transpose+cast this layer's weights to bf16 [N,K]
        tcast(Wq + (size_t)l * E_ * E_,  wq_t, E_, E_);
        tcast(Wk + (size_t)l * E_ * E_,  wk_t, E_, E_);
        tcast(Wv + (size_t)l * E_ * E_,  wv_t, E_, E_);
        tcast(Wo + (size_t)l * E_ * E_,  wo_t, E_, E_);
        tcast(W1 + (size_t)l * E_ * FF_, w1_t, E_, FF_);   // -> [FF, E]
        tcast(W2 + (size_t)l * FF_ * E_, w2_t, FF_, E_);   // -> [E, FF]

        const float* lbq = bq + (size_t)l * E_;
        const float* lbk = bk + (size_t)l * E_;
        const float* lbv = bv + (size_t)l * E_;
        const float* lbo = bo + (size_t)l * E_;
        const float* lb1 = b1 + (size_t)l * FF_;
        const float* lb2 = b2 + (size_t)l * E_;

        // QKV projections -> bf16
        gemm_big(h_bf, E_, 0, wq_t, E_, 0, q_bf, E_, 0,
                 MTOK, E_, E_, 1, 1.0f, lbq, 0, 1);
        gemm_big(h_bf, E_, 0, wk_t, E_, 0, k_bf, E_, 0,
                 MTOK, E_, E_, 1, 1.0f, lbk, 0, 1);
        gemm_big(h_bf, E_, 0, wv_t, E_, 0, v_bf, E_, 0,
                 MTOK, E_, E_, 1, 1.0f, lbv, 0, 1);

        // Per-head V transpose: [96][512][64] -> [96][64][512]
        transpose_v_head<<<dim3(DH_ / 32, S_ / 32, HB), 256, 0, stream>>>(v_bf, vT_bf);

        // scores[g] = scale * q[g] @ k[g]^T  (k already [N=512, K=64])
        gemm_big(q_bf, DH_, HSTR, k_bf, DH_, HSTR, scores, S_, (long long)S_ * S_,
                 S_, S_, DH_, HB, attn_scale, nullptr, 0, 0);

        // row softmax -> bf16
        softmax_rows<<<HB * S_, 256, 0, stream>>>(scores, attn_bf);

        // ctx[g] = attn[g] @ v[g]  (B = vT[g] as [N=64, K=512])
        gemm_sm(attn_bf, S_, (long long)S_ * S_, vT_bf, S_, HSTR,
                ctx_bf, DH_, HSTR, S_, DH_, S_, HB, 1.0f, nullptr, 0, 1);

        // output projection -> f32
        gemm_big(ctx_bf, E_, 0, wo_t, E_, 0, tmp_f32, E_, 0,
                 MTOK, E_, E_, 1, 1.0f, lbo, 0, 0);

        // h = LN(o + h)
        residual_layernorm<<<MTOK, 256, 0, stream>>>(
            tmp_f32, h_f32, h_bf, ln1g + (size_t)l * E_, ln1b + (size_t)l * E_);

        // FFN1 + ReLU -> bf16
        gemm_big(h_bf, E_, 0, w1_t, E_, 0, ffn1_bf, FF_, 0,
                 MTOK, FF_, E_, 1, 1.0f, lb1, 1, 1);

        // FFN2 -> f32
        gemm_big(ffn1_bf, FF_, 0, w2_t, FF_, 0, tmp_f32, E_, 0,
                 MTOK, E_, FF_, 1, 1.0f, lb2, 0, 0);

        // h = LN(ffn + h)
        residual_layernorm<<<MTOK, 256, 0, stream>>>(
            tmp_f32, h_f32, h_bf, ln2g + (size_t)l * E_, ln2b + (size_t)l * E_);
    }

    // Classifier head
    classifier<<<B_, 256, 0, stream>>>(h_f32, Wc, bc, out);
}
